// ConvFFNBlock_61899068670012
// MI455X (gfx1250) — compile-verified, matrix operand order checked by reading, not run on hardware
//
#include <hip/hip_runtime.h>

typedef __bf16 bf16;
typedef __attribute__((ext_vector_type(16))) __bf16 v16bf;
typedef __attribute__((ext_vector_type(8)))  float   v8f;

#define Bsz 8
#define Dd  128
#define Mm  64
#define Nn  1024

// native float -> bf16 convert (hardware v_cvt_pk_bf16_f32 on gfx1250)
__device__ __forceinline__ bf16 f2bf(float f) { return (bf16)f; }

// Branch-free erf (Abramowitz & Stegun 7.1.26, |err| < 1.5e-7):
// one v_rcp_f32 + one v_exp_f32 + Horner poly; no EXEC divergence.
__device__ __forceinline__ float erf_bf(float z) {
    const float a1 =  0.254829592f, a2 = -0.284496736f, a3 = 1.421413741f;
    const float a4 = -1.453152027f, a5 =  1.061405429f;
    float az = fabsf(z);
    float t  = __builtin_amdgcn_rcpf(fmaf(0.3275911f, az, 1.0f));
    float p  = t * fmaf(t, fmaf(t, fmaf(t, fmaf(t, a5, a4), a3), a2), a1);
    float r  = 1.0f - p * __expf(-az * az);
    return copysignf(r, z);
}

__device__ __forceinline__ float gelu_exact(float v) {
    return 0.5f * v * (1.0f + erf_bf(v * 0.70710678118654752f));
}

// ---------------- Weight pre-pack: fp32 (G, S, S) row-major -> bf16 WMMA A-fragments.
// Fragment order: (((g * ksteps + ks) * tiles + t) * 32 + lane) * 16 elements.
// Lane L<16 = row 16t+L, K = ks*32 + {0..7, 16..23}; lanes>=16 get K offset +8.
__global__ __launch_bounds__(256)
void pack_weights(const float* __restrict__ W, bf16* __restrict__ out, int S) {
    const int tid   = (int)(blockIdx.x * blockDim.x + threadIdx.x);
    const int lane  = tid & 31;
    const int tiles = S >> 4, ksteps = S >> 5;
    int tmp = tid >> 5;
    const int t  = tmp % tiles;  tmp /= tiles;
    const int ks = tmp % ksteps; tmp /= ksteps;
    const int g  = tmp;
    const int row = 16 * t + (lane & 15);
    const int kb  = (lane >> 4) << 3;
    const float* wrow = W + ((size_t)g * S + row) * S + ks * 32 + kb;
    bf16* dst = out + (size_t)tid * 16;
#pragma unroll
    for (int e = 0; e < 8; ++e) {
        dst[e]     = f2bf(wrow[e]);
        dst[8 + e] = f2bf(wrow[16 + e]);
    }
}

// ---------------- Kernel 1: FFN1 (D groups of 64x64), writes h as bf16 in (b,m,d,n)
__global__ __launch_bounds__(256)
void ffn1_kernel(const float* __restrict__ x,
                 const v16bf* __restrict__ WAf, const float* __restrict__ b1a,
                 const v16bf* __restrict__ WBf, const float* __restrict__ b1b,
                 bf16* __restrict__ h) {
    const int wave = (int)((blockIdx.x * blockDim.x + threadIdx.x) >> 5);
    const int lane = threadIdx.x & 31;
    const int n0   = (wave & 63) << 4;
    const int d    = (wave >> 6) & (Dd - 1);
    const int b    = wave >> 13;
    const int col  = lane & 15;
    const int kb   = (lane >> 4) << 3;

    const float* xp = x + (((size_t)b * Dd + d) * Mm) * Nn + n0 + col;

    v8f c[4];
#pragma unroll
    for (int t = 0; t < 4; ++t) c[t] = (v8f){};

    // ---- GEMM 1: c = W1a @ x
#pragma unroll
    for (int ks = 0; ks < 2; ++ks) {
        const int k0 = ks * 32;
        v16bf bfrag;
#pragma unroll
        for (int e = 0; e < 16; ++e) {
            const int kk = k0 + kb + (e & 7) + ((e >> 3) << 4);
            bfrag[e] = f2bf(xp[(size_t)kk * Nn]);
        }
#pragma unroll
        for (int t = 0; t < 4; ++t) {
            v16bf af = WAf[(size_t)(((d * 2 + ks) * 4 + t)) * 32 + lane];
            c[t] = __builtin_amdgcn_wmma_f32_16x16x32_bf16(false, af, false, bfrag,
                                                           (short)0, c[t], false, false);
        }
    }
    // bias + GELU (branch-free)
#pragma unroll
    for (int t = 0; t < 4; ++t)
#pragma unroll
        for (int r = 0; r < 8; ++r)
            c[t][r] = gelu_exact(c[t][r] + b1a[d * Mm + 16 * t + kb + r]);

    // ---- GEMM 2: g = W1b @ gelu(...). C-tile layout == B-operand layout.
    v8f g[4];
#pragma unroll
    for (int t = 0; t < 4; ++t) g[t] = (v8f){};
#pragma unroll
    for (int ks = 0; ks < 2; ++ks) {
        const int t0 = ks * 2;
        v16bf bfrag;
#pragma unroll
        for (int e = 0; e < 8; ++e) {
            bfrag[e]     = f2bf(c[t0][e]);
            bfrag[8 + e] = f2bf(c[t0 + 1][e]);
        }
#pragma unroll
        for (int t = 0; t < 4; ++t) {
            v16bf af = WBf[(size_t)(((d * 2 + ks) * 4 + t)) * 32 + lane];
            g[t] = __builtin_amdgcn_wmma_f32_16x16x32_bf16(false, af, false, bfrag,
                                                           (short)0, g[t], false, false);
        }
    }
    // bias + store transposed (b, m, d, n) as bf16
#pragma unroll
    for (int t = 0; t < 4; ++t)
#pragma unroll
        for (int r = 0; r < 8; ++r) {
            const int m = 16 * t + kb + r;
            h[(((size_t)b * Mm + m) * Dd + d) * Nn + n0 + col] = f2bf(g[t][r] + b1b[d * Mm + m]);
        }
}

// ---------------- Kernel 2: FFN2 (M groups of 128x128) + residual, writes (b,d,m,n) fp32
__global__ __launch_bounds__(256)
void ffn2_kernel(const bf16* __restrict__ h, const float* __restrict__ x,
                 const v16bf* __restrict__ WAf, const float* __restrict__ b2a,
                 const v16bf* __restrict__ WBf, const float* __restrict__ b2b,
                 float* __restrict__ out) {
    const int wave = (int)((blockIdx.x * blockDim.x + threadIdx.x) >> 5);
    const int lane = threadIdx.x & 31;
    const int n0   = (wave & 63) << 4;
    const int m    = (wave >> 6) & (Mm - 1);
    const int b    = wave >> 12;
    const int col  = lane & 15;
    const int kb   = (lane >> 4) << 3;

    const bf16* hp = h + (((size_t)b * Mm + m) * Dd) * Nn + n0 + col;

    v8f c[8];
#pragma unroll
    for (int t = 0; t < 8; ++t) c[t] = (v8f){};

    // ---- GEMM 1: c = W2a @ h (K = 128, 4 K-steps; 8 row tiles)
#pragma unroll
    for (int ks = 0; ks < 4; ++ks) {
        const int k0 = ks * 32;
        v16bf bfrag;
#pragma unroll
        for (int e = 0; e < 16; ++e) {
            const int kk = k0 + kb + (e & 7) + ((e >> 3) << 4);
            bfrag[e] = hp[(size_t)kk * Nn];
        }
#pragma unroll
        for (int t = 0; t < 8; ++t) {
            v16bf af = WAf[(size_t)(((m * 4 + ks) * 8 + t)) * 32 + lane];
            c[t] = __builtin_amdgcn_wmma_f32_16x16x32_bf16(false, af, false, bfrag,
                                                           (short)0, c[t], false, false);
        }
    }
#pragma unroll
    for (int t = 0; t < 8; ++t)
#pragma unroll
        for (int r = 0; r < 8; ++r)
            c[t][r] = gelu_exact(c[t][r] + b2a[m * Dd + 16 * t + kb + r]);

    // ---- GEMM 2: g = W2b @ gelu(...) via register-local C->B refeed
    v8f g[8];
#pragma unroll
    for (int t = 0; t < 8; ++t) g[t] = (v8f){};
#pragma unroll
    for (int ks = 0; ks < 4; ++ks) {
        const int t0 = ks * 2;
        v16bf bfrag;
#pragma unroll
        for (int e = 0; e < 8; ++e) {
            bfrag[e]     = f2bf(c[t0][e]);
            bfrag[8 + e] = f2bf(c[t0 + 1][e]);
        }
#pragma unroll
        for (int t = 0; t < 8; ++t) {
            v16bf af = WBf[(size_t)(((m * 4 + ks) * 8 + t)) * 32 + lane];
            g[t] = __builtin_amdgcn_wmma_f32_16x16x32_bf16(false, af, false, bfrag,
                                                           (short)0, g[t], false, false);
        }
    }
    // bias + permute back to (b,d,m,n) + residual
#pragma unroll
    for (int t = 0; t < 8; ++t)
#pragma unroll
        for (int r = 0; r < 8; ++r) {
            const int dd = 16 * t + kb + r;
            const size_t idx = (((size_t)b * Dd + dd) * Mm + m) * Nn + n0 + col;
            out[idx] = x[idx] + g[t][r] + b2b[m * Dd + dd];
        }
}

extern "C" void kernel_launch(void* const* d_in, const int* in_sizes, int n_in,
                              void* d_out, int out_size, void* d_ws, size_t ws_size,
                              hipStream_t stream) {
    const float* x   = (const float*)d_in[0];
    const float* W1a = (const float*)d_in[1];
    const float* b1a = (const float*)d_in[2];
    const float* W1b = (const float*)d_in[3];
    const float* b1b = (const float*)d_in[4];
    const float* W2a = (const float*)d_in[5];
    const float* b2a = (const float*)d_in[6];
    const float* W2b = (const float*)d_in[7];
    const float* b2b = (const float*)d_in[8];
    float* out = (float*)d_out;

    // workspace layout (bf16 elements):
    //   h      : B*M*D*N = 67,108,864 (128 MB)
    //   w1a/w1b: D*M*M   =    524,288 (1 MB each)
    //   w2a/w2b: M*D*D   =  1,048,576 (2 MB each)
    bf16* h    = (bf16*)d_ws;
    bf16* w1af = h    + (size_t)Bsz * Mm * Dd * Nn;
    bf16* w1bf = w1af + (size_t)Dd * Mm * Mm;
    bf16* w2af = w1bf + (size_t)Dd * Mm * Mm;
    bf16* w2bf = w2af + (size_t)Mm * Dd * Dd;

    // pre-pack weights into WMMA A-fragment layout
    pack_weights<<<(Dd * 2 * 4 * 32) / 256, 256, 0, stream>>>(W1a, w1af, Mm);
    pack_weights<<<(Dd * 2 * 4 * 32) / 256, 256, 0, stream>>>(W1b, w1bf, Mm);
    pack_weights<<<(Mm * 4 * 8 * 32) / 256, 256, 0, stream>>>(W2a, w2af, Dd);
    pack_weights<<<(Mm * 4 * 8 * 32) / 256, 256, 0, stream>>>(W2b, w2bf, Dd);

    // kernel 1: one wave per (b, d, 16-col strip) -> 8*128*64 = 65536 waves
    ffn1_kernel<<<65536 / 8, 256, 0, stream>>>(x, (const v16bf*)w1af, b1a,
                                               (const v16bf*)w1bf, b1b, h);
    // kernel 2: one wave per (b, m, 16-col strip) -> 8*64*64 = 32768 waves
    ffn2_kernel<<<32768 / 8, 256, 0, stream>>>(h, x, (const v16bf*)w2af, b2a,
                                               (const v16bf*)w2bf, b2b, out);
}
